// RANGE_30305289240603
// MI455X (gfx1250) — matrix operand order repaired and checked by reading, compiled, not run on hardware
//
#include <hip/hip_runtime.h>
#include <hip/hip_bf16.h>
#include <math.h>

#define N_Q    1024
#define DB_N   100000
#define D_SAT  256
#define D_HR   1024
#define D_OUT  1280
#define TEMP_     12.0f
#define GEO_TEMP_ 40.0f
#define BETA_     0.5f
#define TILE_DB 32
#define NITER   (DB_N / TILE_DB)   /* 3125, exact */
#define NCHUNK  4

typedef __attribute__((ext_vector_type(16))) __bf16 v16bf;
typedef __attribute__((ext_vector_type(8)))  __bf16 v8bf;
typedef __attribute__((ext_vector_type(8)))  float  v8f;
typedef __attribute__((ext_vector_type(2)))  float  v2f;
typedef __attribute__((ext_vector_type(4)))  unsigned int v4u;
typedef __attribute__((ext_vector_type(8)))  int    v8i;
typedef __attribute__((ext_vector_type(4)))  int    v4i;

union AV { v16bf v; v8bf h[2]; };

#if defined(__HIP_DEVICE_COMPILE__) && __has_builtin(__builtin_amdgcn_tensor_load_to_lds) && __has_builtin(__builtin_amdgcn_s_wait_tensorcnt)
#define USE_TDM 1
#endif
#if defined(__HIP_DEVICE_COMPILE__) && __has_builtin(__builtin_amdgcn_wmma_f32_16x16x4_f32)
#define HAS_WMMA_F32X4 1
#endif

#ifdef USE_TDM
// TDM 2D tile load: 32 rows x 256 elements (element size 2B or 4B) -> LDS.
__device__ __forceinline__ void tdm_issue(unsigned int lds_off, const void* gptr, int dsize_code) {
  unsigned long long ga = (unsigned long long)(uintptr_t)gptr;
  v4u g0;
  g0[0] = 1u;                                        // count=1 (valid), no gather
  g0[1] = lds_off;                                   // lds_addr (bytes)
  g0[2] = (unsigned int)ga;                          // global_addr[31:0]
  g0[3] = (unsigned int)((ga >> 32) & 0x01FFFFFFull) | (2u << 30);  // addr[56:32] | type=2
  v8i g1;
  g1[0] = (int)((unsigned)dsize_code << 16);         // wg_mask=0, data_size
  g1[1] = (int)(((unsigned)D_SAT & 0xFFFFu) << 16);  // tensor_dim0 lo16
  g1[2] = (int)(((unsigned)DB_N & 0xFFFFu) << 16);   // tensor_dim1 lo | tensor_dim0 hi(=0)
  g1[3] = (int)(((unsigned)D_SAT & 0xFFFFu) << 16) |
          (int)(((unsigned)DB_N >> 16) & 0xFFFFu);   // tile_dim0=256 | tensor_dim1 hi
  g1[4] = TILE_DB;                                   // tile_dim1=32, tile_dim2=0
  g1[5] = D_SAT;                                     // tensor_dim0_stride lo32
  g1[6] = 0;
  g1[7] = 0;
  v4i g2 = {0,0,0,0};
  v4i g3 = {0,0,0,0};
  v8i g4 = {0,0,0,0,0,0,0,0};
  __builtin_amdgcn_tensor_load_to_lds(g0, g1, g2, g3, g4, 0);
}
#endif

// Online-softmax update; per-lane scalar state for query q = lane&15.
// st0/st1: C-layout S^T logit tiles. Rescale of acc gated on __any(max moved).
// Returns P^T in bf16 B-layout [32db x 16q].
__device__ __forceinline__ v16bf softmax_update(v8f st0, v8f st1, float& m, float& s,
                                                v8f* accA, v8f* accB, int lane) {
  float tm = -INFINITY;
#pragma unroll
  for (int r = 0; r < 8; ++r) { tm = fmaxf(tm, st0[r]); tm = fmaxf(tm, st1[r]); }
  tm = fmaxf(tm, __shfl_xor(tm, 16, 32));
  const float mN = fmaxf(m, tm);
  float e0[8], e1[8], ps = 0.f;
#pragma unroll
  for (int r = 0; r < 8; ++r) {
    e0[r] = __expf(st0[r] - mN); ps += e0[r];
    e1[r] = __expf(st1[r] - mN); ps += e1[r];
  }
  ps += __shfl_xor(ps, 16, 32);
  if (__any(mN > m)) {           // uniform branch; EXEC untouched (WMMA-safe)
    const float resc = __expf(m - mN);
    s = s * resc + ps;
#pragma unroll
    for (int ct = 0; ct < 8; ++ct)
#pragma unroll
      for (int r = 0; r < 8; ++r) { accA[ct][r] *= resc; if (accB) accB[ct][r] *= resc; }
  } else {
    s += ps;
  }
  m = mN;
  // C-layout -> B-layout transpose across lane halves
  v16bf pB;
  const bool lo = lane < 16;
#pragma unroll
  for (int v = 0; v < 4; ++v) {
    float xe = __shfl_xor(e1[2*v],   16, 32);
    float xo = __shfl_xor(e1[2*v+1], 16, 32);
    pB[2*v]   = (__bf16)(lo ? e0[2*v]   : xe);
    pB[2*v+1] = (__bf16)(lo ? e0[2*v+1] : xo);
  }
#pragma unroll
  for (int v = 4; v < 8; ++v) {
    int j = 2*v - 8;
    float xe = __shfl_xor(e0[j],   16, 32);
    float xo = __shfl_xor(e0[j+1], 16, 32);
    pB[2*v]   = (__bf16)(lo ? xe : e1[j]);
    pB[2*v+1] = (__bf16)(lo ? xo : e1[j+1]);
  }
  return pB;
}

// KBF16: K-matrix staged as pre-scaled bf16 (kmat), else f32 dbsat + logit post-scale.
// VT:    V available as bf16 V^T [D_HR][DB_N] (vt), else f32 row-major gather (vhr).
template<bool KBF16, bool VT>
__global__ void __launch_bounds__(256)
range_main(const float* __restrict__ qn,   const float* __restrict__ qx4,
           const float* __restrict__ invn, const float* __restrict__ dbx4,
           const void* __restrict__ kmat,  const float* __restrict__ vhr,
           const __bf16* __restrict__ vt,  float* __restrict__ out,
           float* __restrict__ part, float* __restrict__ msarr) {
  constexpr int TBYTES = TILE_DB * D_SAT * (KBF16 ? 2 : 4);
  __shared__ __align__(16) char smem[2 * TBYTES];
  const int lane  = threadIdx.x & 31;
  const int wave  = threadIdx.x >> 5;
  const int h     = lane >> 4;
  const int qi    = lane & 15;
  const int qbase = blockIdx.x * 16;
  const int cbase = wave * 128;
  const int qrow  = qbase + qi;

  const v8f zz = {0.f,0.f,0.f,0.f,0.f,0.f,0.f,0.f};

  // Resident Q^T in bf16 B-layout: lane: N=qi, K = 32*ks + 16*h + 2v (+1)
  v16bf QB[8];
#pragma unroll
  for (int ks = 0; ks < 8; ++ks)
#pragma unroll
    for (int v = 0; v < 8; ++v) {
      const int k = 32*ks + 16*h + 2*v;
      QB[ks][2*v]   = (__bf16)qn[qrow*D_SAT + k];
      QB[ks][2*v+1] = (__bf16)qn[qrow*D_SAT + k + 1];
    }
  v2f BG; BG.x = qx4[qrow*4 + 2*h]; BG.y = qx4[qrow*4 + 2*h + 1];
#ifndef HAS_WMMA_F32X4
  const float qxs = qx4[qrow*4+0], qys = qx4[qrow*4+1], qzs = qx4[qrow*4+2];
#endif

  v8f accS[8], accG[8];
#pragma unroll
  for (int ct = 0; ct < 8; ++ct) { accS[ct] = zz; accG[ct] = zz; }
  float mS = -INFINITY, sS = 0.f, mG = -INFINITY, sG = 0.f;

  // DB chunk range (flash split-K over grid.y)
  const int itPer = (NITER + gridDim.y - 1) / gridDim.y;
  const int itBeg = blockIdx.y * itPer;
  const int itEnd = (itBeg + itPer < NITER) ? (itBeg + itPer) : NITER;
  constexpr int DSZ = KBF16 ? 1 : 2;     // TDM data_size code: 2B / 4B
  const char* ksrc = (const char*)kmat;
  constexpr size_t KROW = (size_t)D_SAT * (KBF16 ? 2 : 4);

#ifdef USE_TDM
  if (wave == 0) tdm_issue((unsigned int)(uintptr_t)&smem[0], ksrc + (size_t)itBeg*TILE_DB*KROW, DSZ);
#endif

  for (int it = itBeg; it < itEnd; ++it) {
    const int buf = (it - itBeg) & 1;
#ifdef USE_TDM
    if (wave == 0) {
      if (it + 1 < itEnd) {
        tdm_issue((unsigned int)(uintptr_t)&smem[(buf ^ 1) * TBYTES],
                  ksrc + (size_t)(it + 1) * TILE_DB * KROW, DSZ);
        __builtin_amdgcn_s_wait_tensorcnt((short)1);   // current tile done
      } else {
        __builtin_amdgcn_s_wait_tensorcnt((short)0);
      }
    }
    __syncthreads();
#else
    __syncthreads();
    {
      const uint4* src = (const uint4*)(ksrc + (size_t)it * TILE_DB * KROW);
      uint4* dst = (uint4*)&smem[buf * TBYTES];
      for (int idx = threadIdx.x; idx < TBYTES / 16; idx += 256) dst[idx] = src[idx];
    }
    __syncthreads();
#endif

    const int dbb = it * TILE_DB;

    // ---- sem logits: S^T = K~ @ Q^T, two 16x16 tiles, K=256 ----
    v8f st0 = zz, st1 = zz;
    if constexpr (KBF16) {
      const __bf16* Tb = (const __bf16*)&smem[buf * TBYTES];
#pragma unroll
      for (int ks = 0; ks < 8; ++ks) {
        AV a0, a1;
        const __bf16* p0 = Tb + qi * D_SAT + 32*ks + 16*h;
        const __bf16* p1 = Tb + (16 + qi) * D_SAT + 32*ks + 16*h;
        a0.h[0] = *(const v8bf*)(p0);      a0.h[1] = *(const v8bf*)(p0 + 8);
        a1.h[0] = *(const v8bf*)(p1);      a1.h[1] = *(const v8bf*)(p1 + 8);
        st0 = __builtin_amdgcn_wmma_f32_16x16x32_bf16(false, a0.v, false, QB[ks], (short)0, st0, false, false);
        st1 = __builtin_amdgcn_wmma_f32_16x16x32_bf16(false, a1.v, false, QB[ks], (short)0, st1, false, false);
      }
    } else {
      const float* Tf = (const float*)&smem[buf * TBYTES];
#pragma unroll
      for (int ks = 0; ks < 8; ++ks) {
        v16bf a0, a1;
        const float* p0 = Tf + qi * D_SAT + 32*ks + 16*h;
        const float* p1 = Tf + (16 + qi) * D_SAT + 32*ks + 16*h;
#pragma unroll
        for (int j = 0; j < 8; ++j) {
          a0[j]   = (__bf16)p0[j];      a0[8+j] = (__bf16)p0[8 + j];
          a1[j]   = (__bf16)p1[j];      a1[8+j] = (__bf16)p1[8 + j];
        }
        st0 = __builtin_amdgcn_wmma_f32_16x16x32_bf16(false, a0, false, QB[ks], (short)0, st0, false, false);
        st1 = __builtin_amdgcn_wmma_f32_16x16x32_bf16(false, a1, false, QB[ks], (short)0, st1, false, false);
      }
      // post-scale logits by TEMP/||k|| per db row (16 muls instead of 256)
      const float4 sa = *(const float4*)(invn + dbb + 8*h);
      const float4 sb = *(const float4*)(invn + dbb + 8*h + 4);
      const float4 sc = *(const float4*)(invn + dbb + 16 + 8*h);
      const float4 sd = *(const float4*)(invn + dbb + 16 + 8*h + 4);
      st0[0]*=sa.x; st0[1]*=sa.y; st0[2]*=sa.z; st0[3]*=sa.w;
      st0[4]*=sb.x; st0[5]*=sb.y; st0[6]*=sb.z; st0[7]*=sb.w;
      st1[0]*=sc.x; st1[1]*=sc.y; st1[2]*=sc.z; st1[3]*=sc.w;
      st1[4]*=sd.x; st1[5]*=sd.y; st1[6]*=sd.z; st1[7]*=sd.w;
    }

    // ---- geo logits: K=3 matmul (padded to 4) ----
    v8f g0v, g1v;
#ifdef HAS_WMMA_F32X4
    {
      v2f a0g, a1g;
      a0g.x = dbx4[(size_t)(dbb + qi)*4 + 2*h];
      a0g.y = dbx4[(size_t)(dbb + qi)*4 + 2*h + 1];
      a1g.x = dbx4[(size_t)(dbb + 16 + qi)*4 + 2*h];
      a1g.y = dbx4[(size_t)(dbb + 16 + qi)*4 + 2*h + 1];
      g0v = __builtin_amdgcn_wmma_f32_16x16x4_f32(false, a0g, false, BG, (short)0, zz, false, false);
      g1v = __builtin_amdgcn_wmma_f32_16x16x4_f32(false, a1g, false, BG, (short)0, zz, false, false);
    }
#else
#pragma unroll
    for (int r = 0; r < 8; ++r) {
      const int d0 = dbb + 8*h + r, d1 = d0 + 16;
      g0v[r] = dbx4[(size_t)d0*4]*qxs + dbx4[(size_t)d0*4+1]*qys + dbx4[(size_t)d0*4+2]*qzs;
      g1v[r] = dbx4[(size_t)d1*4]*qxs + dbx4[(size_t)d1*4+1]*qys + dbx4[(size_t)d1*4+2]*qzs;
    }
#endif

    const v16bf pBs = softmax_update(st0, st1, mS, sS, accS, nullptr, lane);
    const v16bf pBg = softmax_update(g0v, g1v, mG, sG, accG, nullptr, lane);

    // ---- O^T += V^T @ P^T over this wave's 128-col slice ----
#pragma unroll
    for (int ct = 0; ct < 8; ++ct) {
      AV av;   // A-layout V^T chunk: M=col, K=db row in tile
      if constexpr (VT) {
        const __bf16* vb = vt + (size_t)(cbase + 16*ct + qi) * DB_N + dbb;
        av.h[0] = *(const v8bf*)(vb + 16*h);
        av.h[1] = *(const v8bf*)(vb + 16*h + 8);
      } else {
        const float* Vp = vhr + (size_t)dbb * D_HR + cbase;
        const int c = 16*ct + qi;
#pragma unroll
        for (int j = 0; j < 8; ++j) {
          const int K = 16*h + 2*j;
          av.v[2*j]   = (__bf16)Vp[(size_t)K       * D_HR + c];
          av.v[2*j+1] = (__bf16)Vp[(size_t)(K + 1) * D_HR + c];
        }
      }
      accS[ct] = __builtin_amdgcn_wmma_f32_16x16x32_bf16(false, av.v, false, pBs, (short)0, accS[ct], false, false);
      accG[ct] = __builtin_amdgcn_wmma_f32_16x16x32_bf16(false, av.v, false, pBg, (short)0, accG[ct], false, false);
    }
#ifdef USE_TDM
    __syncthreads();   // reads of buf done before TDM overwrites it (it+2)
#endif
  }

  if (gridDim.y == 1) {
    const float wS = BETA_ / sS;
    const float wG = (1.0f - BETA_) / sG;
#pragma unroll
    for (int ct = 0; ct < 8; ++ct)
#pragma unroll
      for (int r = 0; r < 8; ++r) {
        const int c = cbase + 16*ct + r + 8*h;
        out[(size_t)qrow * D_OUT + c] = accS[ct][r] * wS + accG[ct][r] * wG;
      }
  } else {
    float* pacc = part + ((size_t)blockIdx.y * N_Q + qrow) * (2 * D_HR);
#pragma unroll
    for (int ct = 0; ct < 8; ++ct)
#pragma unroll
      for (int r = 0; r < 8; ++r) {
        const int c = cbase + 16*ct + r + 8*h;
        pacc[c]        = accS[ct][r];
        pacc[D_HR + c] = accG[ct][r];
      }
    if (wave == 0 && h == 0) {
      float* m4 = msarr + ((size_t)blockIdx.y * N_Q + qrow) * 4;
      m4[0] = mS; m4[1] = sS; m4[2] = mG; m4[3] = sG;
    }
  }
}

__global__ void combine_chunks(const float* __restrict__ part, const float* __restrict__ msarr,
                               float* __restrict__ out, int nchunk) {
  const int q = blockIdx.x;
  float mS = -INFINITY, mG = -INFINITY;
  for (int k = 0; k < nchunk; ++k) {
    const float* m4 = msarr + ((size_t)k * N_Q + q) * 4;
    mS = fmaxf(mS, m4[0]); mG = fmaxf(mG, m4[2]);
  }
  float sS = 0.f, sG = 0.f;
  for (int k = 0; k < nchunk; ++k) {
    const float* m4 = msarr + ((size_t)k * N_Q + q) * 4;
    sS += m4[1] * __expf(m4[0] - mS);
    sG += m4[3] * __expf(m4[2] - mG);
  }
  const float wS = BETA_ / sS, wG = (1.0f - BETA_) / sG;
  for (int c = threadIdx.x; c < D_HR; c += 256) {
    float aS = 0.f, aG = 0.f;
    for (int k = 0; k < nchunk; ++k) {
      const float* m4 = msarr + ((size_t)k * N_Q + q) * 4;
      const float* pa = part + ((size_t)k * N_Q + q) * (2 * D_HR);
      aS += pa[c]        * __expf(m4[0] - mS);
      aG += pa[D_HR + c] * __expf(m4[2] - mG);
    }
    out[(size_t)q * D_OUT + c] = aS * wS + aG * wG;
  }
}

__global__ void prep_queries(const float* __restrict__ coords, const float* __restrict__ qe,
                             float* __restrict__ qn, float* __restrict__ qx4,
                             float* __restrict__ out) {
  const int row = blockIdx.x;
  const int t = threadIdx.x;
  __shared__ float red[8];
  __shared__ float rn_s;
  const float x = qe[row * D_SAT + t];
  float ss = x * x;
#pragma unroll
  for (int m = 16; m >= 1; m >>= 1) ss += __shfl_xor(ss, m, 32);
  if ((t & 31) == 0) red[t >> 5] = ss;
  __syncthreads();
  if (t == 0) {
    float tot = 0.f;
#pragma unroll
    for (int i = 0; i < 8; ++i) tot += red[i];
    rn_s = rsqrtf(tot);
    const float d2r = 0.017453292519943295f;
    const float lon = coords[row*2] * d2r, lat = coords[row*2+1] * d2r;
    qx4[row*4+0] = cosf(lat) * cosf(lon);
    qx4[row*4+1] = cosf(lat) * sinf(lon);
    qx4[row*4+2] = sinf(lat);
    qx4[row*4+3] = 0.f;
  }
  __syncthreads();
  const float v = x * rn_s;
  qn[row * D_SAT + t] = v;
  out[(size_t)row * D_OUT + D_HR + t] = v;   // concat q into output tail
}

__global__ void prep_db(const float* __restrict__ dbsat, const float* __restrict__ dbc,
                        float* __restrict__ invn, float* __restrict__ dbx4) {
  const int lane = threadIdx.x & 31;
  const int row = blockIdx.x * 8 + (threadIdx.x >> 5);
  if (row >= DB_N) return;
  float ss = 0.f;
#pragma unroll
  for (int j = 0; j < 8; ++j) {
    const float x = dbsat[(size_t)row * D_SAT + lane + 32*j];
    ss += x * x;
  }
#pragma unroll
  for (int m = 16; m >= 1; m >>= 1) ss += __shfl_xor(ss, m, 32);
  if (lane == 0) {
    invn[row] = TEMP_ * rsqrtf(ss);
    const float d2r = 0.017453292519943295f;
    const float lon = dbc[row*2] * d2r, lat = dbc[row*2+1] * d2r;
    dbx4[(size_t)row*4+0] = GEO_TEMP_ * cosf(lat) * cosf(lon);
    dbx4[(size_t)row*4+1] = GEO_TEMP_ * cosf(lat) * sinf(lon);
    dbx4[(size_t)row*4+2] = GEO_TEMP_ * sinf(lat);
    dbx4[(size_t)row*4+3] = 0.f;
  }
}

// db_satclip -> pre-scaled bf16 rows: kb[row][k] = (TEMP/||row||) * dbsat[row][k]
__global__ void prep_kb(const float* __restrict__ dbsat, __bf16* __restrict__ kb) {
  const int lane = threadIdx.x & 31;
  const int row = blockIdx.x * 8 + (threadIdx.x >> 5);
  if (row >= DB_N) return;
  const float* src = dbsat + (size_t)row * D_SAT + lane * 8;
  float x[8]; float ss = 0.f;
#pragma unroll
  for (int j = 0; j < 8; ++j) { x[j] = src[j]; ss += x[j] * x[j]; }
#pragma unroll
  for (int m = 16; m >= 1; m >>= 1) ss += __shfl_xor(ss, m, 32);
  const float sc = TEMP_ * rsqrtf(ss);
  v8bf o;
#pragma unroll
  for (int j = 0; j < 8; ++j) o[j] = (__bf16)(x[j] * sc);
  *(v8bf*)(kb + (size_t)row * D_SAT + lane * 8) = o;
}

// 32x32 tiled transpose: vhr[DB_N][D_HR] f32 -> vt[D_HR][DB_N] bf16
__global__ void prep_vt(const float* __restrict__ vhr, __bf16* __restrict__ vt) {
  __shared__ float t[32][33];
  const int dbase = blockIdx.x * 32;
  const int cb    = blockIdx.y * 32;
#pragma unroll
  for (int k = 0; k < 4; ++k) {
    const int idx = threadIdx.x + 256 * k;
    const int r = idx >> 5, cc = idx & 31;
    t[r][cc] = vhr[(size_t)(dbase + r) * D_HR + cb + cc];
  }
  __syncthreads();
#pragma unroll
  for (int k = 0; k < 4; ++k) {
    const int idx = threadIdx.x + 256 * k;
    const int c = idx >> 5, dd = idx & 31;
    vt[(size_t)(cb + c) * DB_N + dbase + dd] = (__bf16)t[dd][c];
  }
}

extern "C" void kernel_launch(void* const* d_in, const int* in_sizes, int n_in,
                              void* d_out, int out_size, void* d_ws, size_t ws_size,
                              hipStream_t stream) {
  const float* coords = (const float*)d_in[0];
  const float* qe     = (const float*)d_in[1];
  const float* dbsat  = (const float*)d_in[2];
  const float* dbhr   = (const float*)d_in[3];
  const float* dbc    = (const float*)d_in[4];
  float* out = (float*)d_out;
  char*  ws  = (char*)d_ws;

  // base workspace (always): qn | qx4 | invn | dbx4  (~3.07 MB)
  float* qn   = (float*)ws;
  float* qx4  = qn + N_Q * D_SAT;
  float* invn = qx4 + N_Q * 4;
  float* dbx4 = invn + DB_N;
  size_t off = ((size_t)(N_Q*D_SAT + N_Q*4 + DB_N + DB_N*4)) * 4;

  // optional tiers, runtime-selected from ws_size (deterministic)
  const size_t kbB   = (size_t)DB_N * D_SAT * 2;               // 51.2 MB
  const size_t vtB   = (size_t)DB_N * D_HR * 2;                // 204.8 MB
  const size_t partB = (size_t)NCHUNK * N_Q * (2*D_HR) * 4 +
                       (size_t)NCHUNK * N_Q * 4 * 4;           // ~33.6 MB
  const bool useKB = ws_size >= off + kbB;
  __bf16* kb = (__bf16*)(ws + off); if (useKB) off += kbB;
  const bool useVT = useKB && (ws_size >= off + vtB);
  __bf16* vt = (__bf16*)(ws + off); if (useVT) off += vtB;
  const bool split = ws_size >= off + partB;
  float* part  = (float*)(ws + off);
  float* msarr = part + (size_t)NCHUNK * N_Q * (2*D_HR);
  const int nchunk = split ? NCHUNK : 1;

  prep_queries<<<N_Q, 256, 0, stream>>>(coords, qe, qn, qx4, out);
  prep_db<<<(DB_N + 7) / 8, 256, 0, stream>>>(dbsat, dbc, invn, dbx4);
  if (useKB) prep_kb<<<(DB_N + 7) / 8, 256, 0, stream>>>(dbsat, kb);
  if (useVT) prep_vt<<<dim3(DB_N / 32, D_HR / 32), 256, 0, stream>>>(dbhr, vt);

  const dim3 grid(N_Q / 16, nchunk);
  if (useKB && useVT)
    range_main<true, true><<<grid, 256, 0, stream>>>(qn, qx4, invn, dbx4, kb, dbhr, vt, out, part, msarr);
  else if (useKB)
    range_main<true, false><<<grid, 256, 0, stream>>>(qn, qx4, invn, dbx4, kb, dbhr, vt, out, part, msarr);
  else
    range_main<false, false><<<grid, 256, 0, stream>>>(qn, qx4, invn, dbx4, dbsat, dbhr, vt, out, part, msarr);

  if (split) combine_chunks<<<N_Q, 256, 0, stream>>>(part, msarr, out, nchunk);
}
